// MMMD_9423158247459
// MI455X (gfx1250) — matrix operand order repaired and checked by reading, not compiled, not run on hardware
//
#include <hip/hip_runtime.h>
#include <math.h>

typedef _Float16 f16;
typedef __attribute__((ext_vector_type(16))) _Float16 v16h;
typedef __attribute__((ext_vector_type(8)))  _Float16 v8h;
typedef __attribute__((ext_vector_type(4)))  _Float16 v4h;
typedef __attribute__((ext_vector_type(8)))  float    v8f;
typedef __attribute__((ext_vector_type(4)))  float    v4f;

#define NSRC 4
#define NB   8192
#define ND   128
#define BD   ((size_t)NB * ND)

// mmd tile config
#define MT   128          // M rows per block
#define NT   64           // N cols per block
#define LDST 136          // LDS row stride in f16 (pad: conflict-free b128 reads)

// ---------------------------------------------------------------------------
// prep: fp32 -> fp16 rows, per-row squared norms, zero accumulator.
// One wave (32 lanes) per row; 4 floats per lane.
// ---------------------------------------------------------------------------
__global__ __launch_bounds__(256)
void prep_kernel(const float* __restrict__ src, const float* __restrict__ tgt,
                 f16* __restrict__ hsrc, f16* __restrict__ htgt,
                 float* __restrict__ sqs, float* __restrict__ sqt,
                 float* __restrict__ accum) {
    if (blockIdx.x == 0 && threadIdx.x == 0) accum[0] = 0.0f;

    const int gwave = (int)((blockIdx.x * blockDim.x + threadIdx.x) >> 5);
    const int lane  = threadIdx.x & 31;
    const int R     = NSRC * NB + NB;   // 40960 rows total
    if (gwave >= R) return;

    const float* rp; f16* hp; float* sq;
    if (gwave < NSRC * NB) {
        rp = src  + (size_t)gwave * ND;
        hp = hsrc + (size_t)gwave * ND;
        sq = sqs + gwave;
    } else {
        const int r = gwave - NSRC * NB;
        rp = tgt  + (size_t)r * ND;
        hp = htgt + (size_t)r * ND;
        sq = sqt + r;
    }

    v4f x = *(const v4f*)(rp + lane * 4);
    float s2 = x.x * x.x + x.y * x.y + x.z * x.z + x.w * x.w;

    v4h h;
    h[0] = (f16)x.x; h[1] = (f16)x.y; h[2] = (f16)x.z; h[3] = (f16)x.w;
    *(v4h*)(hp + lane * 4) = h;

    #pragma unroll
    for (int d = 16; d > 0; d >>= 1) s2 += __shfl_down(s2, d, 32);
    if (lane == 0) *sq = s2;
}

// ---------------------------------------------------------------------------
// centers: column sums of each of the 5 matrices (4 sources + target).
// One block per matrix, one thread per feature dim.
// ---------------------------------------------------------------------------
__global__ __launch_bounds__(128)
void centers_kernel(const float* __restrict__ src, const float* __restrict__ tgt,
                    float* __restrict__ centers) {
    const int mat = blockIdx.x;     // 0..4
    const int d   = threadIdx.x;    // 0..127
    const float* base = (mat < NSRC) ? (src + (size_t)mat * BD) : tgt;
    float s = 0.0f;
    for (int b = 0; b < NB; ++b) s += base[(size_t)b * ND + d];
    centers[mat * ND + d] = s;
}

// ---------------------------------------------------------------------------
// mmd: fused GEMM(X Y^T) + gaussian kernel + mean, per pair.
// blockIdx.y = pair p (0..8); blockIdx.x = tile (64 M-tiles x 128 N-tiles).
// 256 threads = 8 wave32; each wave owns one 16-row strip of the 128x64 tile.
// ---------------------------------------------------------------------------
__global__ __launch_bounds__(256)
void mmd_kernel(const f16* __restrict__ hsrc, const f16* __restrict__ htgt,
                const float* __restrict__ sqs, const float* __restrict__ sqt,
                float* __restrict__ accum) {
    __shared__ f16 Xs[MT * LDST];
    __shared__ f16 Ys[NT * LDST];
    __shared__ float red[256];

    const int p      = blockIdx.y;
    const int tm     = blockIdx.x & 63;     // 64 M tiles
    const int tn     = blockIdx.x >> 6;     // 128 N tiles
    const int blockM = tm * MT;
    const int blockN = tn * NT;

    const float invB2 = 1.0f / ((float)NB * (float)NB);
    const f16 *X, *Y; const float *sqx, *sqy; float wgt;
    if (p < NSRC) {                 // K(s,s): +1/(S*B^2)
        X = hsrc + (size_t)p * BD; sqx = sqs + p * NB;
        Y = X;                      sqy = sqx;
        wgt = 0.25f * invB2;
    } else if (p < 2 * NSRC) {      // K(s,t): -2/(S*B^2)
        const int s = p - NSRC;
        X = hsrc + (size_t)s * BD;  sqx = sqs + s * NB;
        Y = htgt;                   sqy = sqt;
        wgt = -0.5f * invB2;
    } else {                        // K(t,t): +1/B^2
        X = htgt; sqx = sqt; Y = htgt; sqy = sqt;
        wgt = invB2;
    }

    const int tid = threadIdx.x;

    // Stage tiles into LDS (16-byte chunks, fully coalesced).
    for (int c = tid; c < MT * 16; c += 256) {
        const int r = c >> 4, col = (c & 15) * 8;
        *(v8h*)&Xs[r * LDST + col] = *(const v8h*)(X + (size_t)(blockM + r) * ND + col);
    }
    for (int c = tid; c < NT * 16; c += 256) {
        const int r = c >> 4, col = (c & 15) * 8;
        *(v8h*)&Ys[r * LDST + col] = *(const v8h*)(Y + (size_t)(blockN + r) * ND + col);
    }
    __syncthreads();

    const int w    = tid >> 5;
    const int lane = tid & 31;
    const int half = lane >> 4;
    const int l16  = lane & 15;

    // A fragments (16x32 f16 layout, ISA 7.12.2):
    // lanes 0-15: row=l16, K{k0..k0+7, k0+16..k0+23}; lanes 16-31: +8 shift.
    v16h A[4];
    #pragma unroll
    for (int k = 0; k < 4; ++k) {
        const int k0  = k * 32;
        const int row = w * 16 + l16;
        v8h c0 = *(const v8h*)&Xs[row * LDST + k0 + half * 8];
        v8h c1 = *(const v8h*)&Xs[row * LDST + k0 + 16 + half * 8];
        A[k] = __builtin_shufflevector(c0, c1, 0,1,2,3,4,5,6,7,8,9,10,11,12,13,14,15);
    }

    // Row squared norms for this wave's strip: C element (v, half) -> M = v + 8*half.
    float sqm[8];
    #pragma unroll
    for (int v = 0; v < 8; ++v) sqm[v] = sqx[blockM + w * 16 + v + 8 * half];

    float lsum = 0.0f;
    #pragma unroll
    for (int n = 0; n < 4; ++n) {
        v8f acc = {0.f, 0.f, 0.f, 0.f, 0.f, 0.f, 0.f, 0.f};
        #pragma unroll
        for (int k = 0; k < 4; ++k) {
            const int k0  = k * 32;
            const int row = n * 16 + l16;   // column N of B = row of Y
            // B fragment (32x16): lanes 0-15 hold K{k0..k0+7, k0+16..k0+23}; lanes 16-31: +8 shift
            v8h c0 = *(const v8h*)&Ys[row * LDST + k0 + half * 8];
            v8h c1 = *(const v8h*)&Ys[row * LDST + k0 + 16 + half * 8];
            v16h Bf = __builtin_shufflevector(c0, c1, 0,1,2,3,4,5,6,7,8,9,10,11,12,13,14,15);
            acc = __builtin_amdgcn_wmma_f32_16x16x32_f16(
                false, A[k], false, Bf, (short)0, acc, false, false);
        }
        const float sqn = sqy[blockN + n * 16 + l16];
        #pragma unroll
        for (int v = 0; v < 8; ++v)
            lsum += __expf(-0.5f * (sqm[v] + sqn - 2.0f * acc[v]));
    }

    red[tid] = lsum;
    __syncthreads();
    #pragma unroll
    for (int s = 128; s > 0; s >>= 1) {
        if (tid < s) red[tid] += red[tid + s];
        __syncthreads();
    }
    if (tid == 0) atomicAdd(accum, wgt * red[0]);
}

// ---------------------------------------------------------------------------
// finalize: cosine penalty from centroid sums, write scalar result.
// ---------------------------------------------------------------------------
__global__ __launch_bounds__(128)
void finalize_kernel(const float* __restrict__ centers,
                     const float* __restrict__ accum,
                     float* __restrict__ out) {
    __shared__ float sh[128];
    const int d = threadIdx.x;
    const float invB = 1.0f / (float)NB;

    const float ct = centers[NSRC * ND + d] * invB;
    float nt2_p = ct * ct;
    float dots_p[NSRC], ns2_p[NSRC];
    #pragma unroll
    for (int s = 0; s < NSRC; ++s) {
        const float cs = centers[s * ND + d] * invB;
        dots_p[s] = cs * ct;
        ns2_p[s]  = cs * cs;
    }

    auto block_reduce = [&](float v) -> float {
        sh[d] = v; __syncthreads();
        for (int k = 64; k > 0; k >>= 1) {
            if (d < k) sh[d] += sh[d + k];
            __syncthreads();
        }
        const float r = sh[0]; __syncthreads();
        return r;
    };

    const float nt2 = block_reduce(nt2_p);
    const float nt  = fmaxf(sqrtf(nt2), 1e-8f);
    float pen = 0.0f;
    for (int s = 0; s < NSRC; ++s) {
        const float dot = block_reduce(dots_p[s]);
        const float ns2 = block_reduce(ns2_p[s]);
        const float ns  = fmaxf(sqrtf(ns2), 1e-8f);
        pen += dot / (ns * nt);
    }
    if (d == 0) out[0] = accum[0] - 0.25f * pen;
}

// ---------------------------------------------------------------------------
extern "C" void kernel_launch(void* const* d_in, const int* in_sizes, int n_in,
                              void* d_out, int out_size, void* d_ws, size_t ws_size,
                              hipStream_t stream) {
    const float* src = (const float*)d_in[0];   // [4, 8192, 128] fp32
    const float* tgt = (const float*)d_in[1];   // [8192, 128] fp32

    char* ws = (char*)d_ws;
    // workspace layout (≈10.65 MB)
    f16*   hsrc    = (f16*)(ws);                             // 4*8192*128 f16 = 8 MiB
    f16*   htgt    = (f16*)(ws + 8388608);                   // 8192*128 f16   = 2 MiB
    float* sqs     = (float*)(ws + 8388608 + 2097152);       // 32768 f32
    float* sqt     = (float*)(ws + 8388608 + 2097152 + 131072);          // 8192 f32
    float* centers = (float*)(ws + 8388608 + 2097152 + 131072 + 32768);  // 640 f32
    float* accum   = (float*)(ws + 8388608 + 2097152 + 131072 + 32768 + 2560); // 1 f32

    // 1) convert + row norms + zero accumulator: 40960 rows, 8 waves/block
    prep_kernel<<<5120, 256, 0, stream>>>(src, tgt, hsrc, htgt, sqs, sqt, accum);

    // 2) centroid column sums
    centers_kernel<<<5, 128, 0, stream>>>(src, tgt, centers);

    // 3) fused WMMA gaussian-kernel mean: 64 M-tiles * 128 N-tiles, 9 pairs
    mmd_kernel<<<dim3(64 * 128, 9), 256, 0, stream>>>(hsrc, htgt, sqs, sqt, accum);

    // 4) cosine penalty + final scalar
    finalize_kernel<<<1, 128, 0, stream>>>(centers, accum, (float*)d_out);
}
